// InfluenceGNN_52063593562729
// MI455X (gfx1250) — matrix operand order repaired and checked by reading, not compiled, not run on hardware
//
#include <hip/hip_runtime.h>
#include <hip/hip_bf16.h>

// ---------------------------------------------------------------------------
// Types for WMMA fragments (trivial ext-vector types so they can live in a
// union; HIP's uint4 has ctors and cannot).
// ---------------------------------------------------------------------------
typedef __attribute__((ext_vector_type(16))) __bf16       v16bf;
typedef __attribute__((ext_vector_type(8)))  float        v8f;
typedef __attribute__((ext_vector_type(4)))  unsigned int u32x4;
typedef __attribute__((ext_vector_type(2)))  unsigned int u32x2;

union ABfrag { u32x4 u[2]; v16bf v; };

__device__ __forceinline__ float bf_lo(unsigned int u) {
  union { unsigned int i; float f; } c; c.i = u << 16; return c.f;
}
__device__ __forceinline__ float bf_hi(unsigned int u) {
  union { unsigned int i; float f; } c; c.i = u & 0xffff0000u; return c.f;
}

#define HIDC 128

// ---------------------------------------------------------------------------
// Pre-pass kernels
// ---------------------------------------------------------------------------
__global__ void k_init_deg(float* deg, int n) {
  int i = blockIdx.x * blockDim.x + threadIdx.x;
  if (i < n) deg[i] = 1.0f;  // self-loop weight
}

__global__ void k_deg_accum(const int* __restrict__ ei, const float* __restrict__ ea,
                            float* __restrict__ deg, int E) {
  int e = blockIdx.x * blockDim.x + threadIdx.x;
  if (e >= E) return;
  unsafeAtomicAdd(&deg[ei[E + e]], ea[e]);  // in-degree by weight (col index)
}

__global__ void k_rsqrt(float* deg, int n) {
  int i = blockIdx.x * blockDim.x + threadIdx.x;
  if (i < n) {
    float d = deg[i];
    deg[i] = d > 0.0f ? rsqrtf(d) : 0.0f;   // becomes dinv
  }
}

__global__ void k_edge_prep(const int* __restrict__ ei, const float* __restrict__ ea,
                            const float* __restrict__ dinv,
                            int* __restrict__ row, int* __restrict__ col,
                            float* __restrict__ norm, int E) {
  int e = blockIdx.x * blockDim.x + threadIdx.x;
  if (e >= E) return;
  int r = ei[e];
  int c = ei[E + e];
  row[e] = r;
  col[e] = c;
  norm[e] = dinv[r] * ea[e] * dinv[c];
}

__global__ void k_cvt_bf16(const float* __restrict__ src, __bf16* __restrict__ dst, int n) {
  int i = blockIdx.x * blockDim.x + threadIdx.x;
  if (i < n) dst[i] = (__bf16)src[i];
}

// W is [K=128][N=128] row-major f32 (x @ W). Store transposed bf16: Wt[n][k].
__global__ void k_wprep(const float* __restrict__ W, __bf16* __restrict__ Wt) {
  int i = blockIdx.x * blockDim.x + threadIdx.x;
  if (i >= HIDC * HIDC) return;
  int k = i >> 7;
  int n = i & 127;
  Wt[n * HIDC + k] = (__bf16)W[i];
}

__global__ void k_zero(float* p, int n) {
  int i = blockIdx.x * blockDim.x + threadIdx.x;
  if (i < n) p[i] = 0.0f;
}

// ---------------------------------------------------------------------------
// GEMM: xw = h @ W  via v_wmma_f32_16x16x32_bf16.
// 8 waves / block. The 32KB transposed weight is async-staged into LDS once
// per block (global_load_async_to_lds_b128, ASYNCcnt path), then each wave
// computes TPW consecutive 16-row strips reading B fragments from LDS.
// ---------------------------------------------------------------------------
#define GEMM_WAVES 8
#define TPW 4   // tiles per wave

__global__ void __launch_bounds__(GEMM_WAVES * 32)
k_gemm_bf16(const __bf16* __restrict__ A, const __bf16* __restrict__ Bt,
            __bf16* __restrict__ C, int nTiles, int nRows) {
  __shared__ __bf16 sB[HIDC * HIDC];            // 32 KB (of 320 KB/WGP)

  const int tid  = threadIdx.x;                 // 0..255
  const int wave = tid >> 5;
  const int lane = tid & 31;

  // ---- async stage Bt -> LDS: 256 threads x 8 iters x 16B = 32KB ----
#pragma unroll
  for (int i = 0; i < (HIDC * HIDC) / (GEMM_WAVES * 32 * 8); ++i) {
    int elem = (tid + i * GEMM_WAVES * 32) * 8;         // bf16 elements (16B chunk)
    unsigned      ldsAddr = (unsigned)(uintptr_t)(&sB[elem]);  // low 32b = LDS offset
    const __bf16* gsrc    = Bt + elem;
    asm volatile("global_load_async_to_lds_b128 %0, %1, off"
                 :: "v"(ldsAddr), "v"(gsrc) : "memory");
  }
  asm volatile("s_wait_asynccnt 0x0" ::: "memory");
  __syncthreads();

  const int m    = lane & 15;           // row within A tile / column within B tile
  const int half = lane >> 4;           // K sub-block selector per ISA layout
  const int baseTile = blockIdx.x * (GEMM_WAVES * TPW) + wave * TPW;

  for (int t = 0; t < TPW; ++t) {
    const int tile = baseTile + t;      // wave-uniform guard: EXEC all-1 inside
    if (tile >= nTiles) break;

    int arow = tile * 16 + m;
    if (arow >= nRows) arow = nRows - 1;  // clamp (N is a multiple of 16 here)
    const __bf16* aptr = A + (size_t)arow * HIDC + half * 8;

    v8f acc[8];
#pragma unroll
    for (int nt = 0; nt < 8; ++nt) acc[nt] = (v8f){0.f,0.f,0.f,0.f,0.f,0.f,0.f,0.f};

#pragma unroll
    for (int k0 = 0; k0 < HIDC; k0 += 32) {
      ABfrag a;
      a.u[0] = *(const u32x4*)(aptr + k0);        // K = k0 + 8*half + [0..7]
      a.u[1] = *(const u32x4*)(aptr + k0 + 16);   // K = k0 + 8*half + [16..23]
#pragma unroll
      for (int nt = 0; nt < 8; ++nt) {
        const __bf16* bptr = &sB[(nt * 16 + m) * HIDC + k0 + half * 8];
        ABfrag b;
        b.u[0] = *(const u32x4*)(bptr);           // ds_load_b128: K = k0+8*half+[0..7]
        b.u[1] = *(const u32x4*)(bptr + 16);      //               K = k0+8*half+[16..23]
        acc[nt] = __builtin_amdgcn_wmma_f32_16x16x32_bf16(
            false, a.v, false, b.v, (short)0, acc[nt], false, false);
      }
    }

    // D layout: lane l holds column n = l&15; VGPR v holds row v + 8*(l>=16).
#pragma unroll
    for (int nt = 0; nt < 8; ++nt) {
#pragma unroll
      for (int v = 0; v < 8; ++v) {
        int r = tile * 16 + v + 8 * half;
        if (r < nRows) C[(size_t)r * HIDC + nt * 16 + m] = (__bf16)acc[nt][v];
      }
    }
  }
}

// ---------------------------------------------------------------------------
// Scatter: one wave per edge; lane handles 4 features.
// agg[col] += norm * xw[row]   (bf16 gather, f32 native atomic accumulate)
// ---------------------------------------------------------------------------
__global__ void k_scatter(const int* __restrict__ row, const int* __restrict__ col,
                          const float* __restrict__ norm, const __bf16* __restrict__ xw,
                          float* __restrict__ agg, int E) {
  int wid  = (blockIdx.x * blockDim.x + threadIdx.x) >> 5;
  int lane = threadIdx.x & 31;
  if (wid >= E) return;
  int   r  = row[wid];
  int   c  = col[wid];
  float nm = norm[wid];
  u32x2 p  = *(const u32x2*)(xw + (size_t)r * HIDC + lane * 4);
  float* dst = agg + (size_t)c * HIDC + lane * 4;
  unsafeAtomicAdd(dst + 0, nm * bf_lo(p.x));
  unsafeAtomicAdd(dst + 1, nm * bf_hi(p.x));
  unsafeAtomicAdd(dst + 2, nm * bf_lo(p.y));
  unsafeAtomicAdd(dst + 3, nm * bf_hi(p.y));
}

// ---------------------------------------------------------------------------
// Finalize: add self-loop term dinv^2 * xw, bias, ReLU -> bf16 activations.
// ---------------------------------------------------------------------------
__global__ void k_finalize(const float* __restrict__ agg, const __bf16* __restrict__ xw,
                           const float* __restrict__ dinv, const float* __restrict__ bias,
                           __bf16* __restrict__ hout, int total) {
  int i = blockIdx.x * blockDim.x + threadIdx.x;
  if (i >= total) return;
  int node = i >> 7;
  int c    = i & 127;
  float di = dinv[node];
  float v  = agg[i] + di * di * (float)xw[i] + bias[c];
  hout[i]  = (__bf16)(v > 0.0f ? v : 0.0f);
}

// ---------------------------------------------------------------------------
// Final linear 128 -> 1: wave-per-node dot + wave32 shuffle reduce.
// ---------------------------------------------------------------------------
__global__ void k_out_linear(const __bf16* __restrict__ h, const float* __restrict__ w,
                             const float* __restrict__ b, float* __restrict__ out, int N) {
  int wid  = (blockIdx.x * blockDim.x + threadIdx.x) >> 5;
  int lane = threadIdx.x & 31;
  if (wid >= N) return;
  u32x2 p = *(const u32x2*)(h + (size_t)wid * HIDC + lane * 4);
  const float* wp = w + lane * 4;
  float s = bf_lo(p.x) * wp[0] + bf_hi(p.x) * wp[1] +
            bf_lo(p.y) * wp[2] + bf_hi(p.y) * wp[3];
#pragma unroll
  for (int off = 16; off > 0; off >>= 1) s += __shfl_down(s, off, 32);
  if (lane == 0) out[wid] = s + b[0];
}

// ---------------------------------------------------------------------------
// Host-side orchestration
// ---------------------------------------------------------------------------
extern "C" void kernel_launch(void* const* d_in, const int* in_sizes, int n_in,
                              void* d_out, int out_size, void* d_ws, size_t ws_size,
                              hipStream_t stream) {
  const float* x   = (const float*)d_in[0];
  const int*   ei  = (const int*)d_in[1];     // jax int64 w/o x64 => int32
  const float* ea  = (const float*)d_in[2];
  const float* Wf[3]   = {(const float*)d_in[3], (const float*)d_in[5], (const float*)d_in[7]};
  const float* bias[3] = {(const float*)d_in[4], (const float*)d_in[6], (const float*)d_in[8]};
  const float* lw  = (const float*)d_in[9];
  const float* lb  = (const float*)d_in[10];

  const int N = in_sizes[0] / HIDC;
  const int E = in_sizes[1] / 2;

  // Workspace carve-out (256B aligned slices)
  char*  ws  = (char*)d_ws;
  size_t off = 0;
  auto carve = [&](size_t bytes) -> char* {
    char* p = ws + off;
    off += (bytes + 255u) & ~(size_t)255u;
    return p;
  };
  float*  dinv   = (float*)carve((size_t)N * 4);                 // deg -> dinv
  int*    row32  = (int*)carve((size_t)E * 4);
  int*    col32  = (int*)carve((size_t)E * 4);
  float*  norm   = (float*)carve((size_t)E * 4);
  __bf16* Wt     = (__bf16*)carve((size_t)3 * HIDC * HIDC * 2);  // 3 transposed weights
  __bf16* hbuf   = (__bf16*)carve((size_t)N * HIDC * 2);         // activations
  __bf16* xwbuf  = (__bf16*)carve((size_t)N * HIDC * 2);         // h @ W
  float*  agg    = (float*)carve((size_t)N * HIDC * 4);          // scatter accumulator
  (void)ws_size;

  const int TB = 256;
  const int NH = N * HIDC;

  // ---- normalization pre-pass ----
  k_init_deg<<<(N + TB - 1) / TB, TB, 0, stream>>>(dinv, N);
  k_deg_accum<<<(E + TB - 1) / TB, TB, 0, stream>>>(ei, ea, dinv, E);
  k_rsqrt<<<(N + TB - 1) / TB, TB, 0, stream>>>(dinv, N);
  k_edge_prep<<<(E + TB - 1) / TB, TB, 0, stream>>>(ei, ea, dinv, row32, col32, norm, E);

  // ---- precision prep: bf16 activations + transposed bf16 weights ----
  k_cvt_bf16<<<(NH + TB - 1) / TB, TB, 0, stream>>>(x, hbuf, NH);
  for (int l = 0; l < 3; ++l)
    k_wprep<<<(HIDC * HIDC + TB - 1) / TB, TB, 0, stream>>>(Wf[l], Wt + (size_t)l * HIDC * HIDC);

  // ---- 3 GCN layers ----
  const int nTiles     = (N + 15) / 16;
  const int tilesPerBlk = GEMM_WAVES * TPW;        // 32 tiles (512 rows) per block
  const int gemmGrid   = (nTiles + tilesPerBlk - 1) / tilesPerBlk;
  const int scatGrid   = (E + 7) / 8;              // 8 waves (edges) per block

  for (int l = 0; l < 3; ++l) {
    k_zero<<<(NH + TB - 1) / TB, TB, 0, stream>>>(agg, NH);
    k_gemm_bf16<<<gemmGrid, GEMM_WAVES * 32, 0, stream>>>(
        hbuf, Wt + (size_t)l * HIDC * HIDC, xwbuf, nTiles, N);
    k_scatter<<<scatGrid, TB, 0, stream>>>(row32, col32, norm, xwbuf, agg, E);
    k_finalize<<<(NH + TB - 1) / TB, TB, 0, stream>>>(agg, xwbuf, dinv, bias[l], hbuf, NH);
  }

  // ---- final linear 128 -> 1 ----
  k_out_linear<<<(N + 7) / 8, TB, 0, stream>>>(hbuf, lw, lb, (float*)d_out, N);
}
